// MhAttention_58334245814274
// MI455X (gfx1250) — matrix operand order repaired and checked by reading, not compiled, not run on hardware
//
#include <hip/hip_runtime.h>
#include <hip/hip_bf16.h>

typedef _Float16 f16;
typedef __attribute__((ext_vector_type(16))) _Float16 v16h;
typedef __attribute__((ext_vector_type(8)))  _Float16 v8h;
typedef __attribute__((ext_vector_type(8)))  float    v8f;

#define D_MODEL 1024
#define N_HEADS 16
#define BATCH   2
#define SEQ     2048
#define HD      64          // head dim
#define MROWS   (BATCH*SEQ) // 4096
#define GK      D_MODEL     // GEMM K
#define GN      D_MODEL     // GEMM N

__device__ __forceinline__ v16h mk16(v8h lo, v8h hi) {
    v16h r;
#pragma unroll
    for (int e = 0; e < 8; ++e) { r[e] = lo[e]; r[8 + e] = hi[e]; }
    return r;
}

// ---------------------------------------------------------------- converts
__global__ void cvt_f16_kernel(const float* __restrict__ src, f16* __restrict__ dst, int n) {
    int i = blockIdx.x * blockDim.x + threadIdx.x;
    if (i < n) dst[i] = (f16)src[i];
}

// w[k][n] f32 -> wT[n][k] f16  (so WMMA B-operand reads are K-contiguous)
__global__ void cvt_wT_kernel(const float* __restrict__ w, f16* __restrict__ wT) {
    int i = blockIdx.x * blockDim.x + threadIdx.x; // D*D threads
    int k = i >> 10, n = i & 1023;
    wT[n * D_MODEL + k] = (f16)w[k * D_MODEL + n];
}

// ---------------------------------------------------------------- WMMA GEMM
// C[4096,1024] = A[4096,1024] (f16 row-major) * BT[1024,1024] (f16, B transposed)
// block tile 128x256, 8 waves of 64x64, K-step 32, double-buffered LDS
#define LDT 40
template <int STORE_HALF>
__global__ __launch_bounds__(256) void gemm_wmma(const f16* __restrict__ A,
                                                 const f16* __restrict__ BT,
                                                 float* __restrict__ Cf,
                                                 f16*  __restrict__ Ch) {
    __shared__ f16 As[2][128 * LDT];
    __shared__ f16 Bs[2][256 * LDT];
    const int t = threadIdx.x;
    const int lane = t & 31, wave = t >> 5;
    const int wm = wave & 1, wn = wave >> 1;
    const int gm0 = blockIdx.y * 128, gn0 = blockIdx.x * 256;
    const int ln = lane & 15;
    const int hb = (lane < 16) ? 0 : 8;   // C-row offset & A kb8
    const int kb16 = (lane < 16) ? 0 : 16;

    const int rowA = t >> 1, segA = (t & 1) * 16;
    const f16* gA = A + (long)(gm0 + rowA) * GK + segA;
    const f16* gB = BT + (long)(gn0 + t) * GK;
    f16* sA = &As[0][rowA * LDT + segA];
    f16* sB = &Bs[0][t * LDT];
    const int bufStride_A = 128 * LDT;
    const int bufStride_B = 256 * LDT;

    // preload tile 0
    {
        v8h a0 = *(const v8h*)gA;
        v8h a1 = *(const v8h*)(gA + 8);
        v8h b0 = *(const v8h*)gB;       v8h b1 = *(const v8h*)(gB + 8);
        v8h b2 = *(const v8h*)(gB + 16); v8h b3 = *(const v8h*)(gB + 24);
        *(v8h*)sA = a0; *(v8h*)(sA + 8) = a1;
        *(v8h*)sB = b0; *(v8h*)(sB + 8) = b1; *(v8h*)(sB + 16) = b2; *(v8h*)(sB + 24) = b3;
    }
    __syncthreads();

    v8f acc[4][4] = {};
    int cur = 0;
    for (int kk = 0; kk < GK; kk += 32) {
        // issue next tile's global loads (overlap with WMMA below)
        v8h pa0, pa1, pb0, pb1, pb2, pb3;
        const bool more = (kk + 32) < GK;
        if (more) {
            const f16* nA = gA + kk + 32;
            const f16* nB = gB + kk + 32;
            pa0 = *(const v8h*)nA;        pa1 = *(const v8h*)(nA + 8);
            pb0 = *(const v8h*)nB;        pb1 = *(const v8h*)(nB + 8);
            pb2 = *(const v8h*)(nB + 16); pb3 = *(const v8h*)(nB + 24);
        }

        // compute on current buffer
        const f16* Ab = &As[cur][0];
        const f16* Bb = &Bs[cur][0];
        v16h a[4];
#pragma unroll
        for (int mi = 0; mi < 4; ++mi) {
            const f16* p = &Ab[(wm * 64 + mi * 16 + ln) * LDT + hb];
            a[mi] = mk16(*(const v8h*)p, *(const v8h*)(p + 16));
        }
#pragma unroll
        for (int ni = 0; ni < 4; ++ni) {
            const f16* p = &Bb[(wn * 64 + ni * 16 + ln) * LDT + hb];
            v16h bf = mk16(*(const v8h*)p, *(const v8h*)(p + 16));
#pragma unroll
            for (int mi = 0; mi < 4; ++mi)
                acc[mi][ni] = __builtin_amdgcn_wmma_f32_16x16x32_f16(
                    false, a[mi], false, bf, (short)0, acc[mi][ni], false, false);
        }

        if (more) {
            f16* dA = sA + (cur ^ 1) * bufStride_A;
            f16* dB = sB + (cur ^ 1) * bufStride_B;
            *(v8h*)dA = pa0; *(v8h*)(dA + 8) = pa1;
            *(v8h*)dB = pb0; *(v8h*)(dB + 8) = pb1;
            *(v8h*)(dB + 16) = pb2; *(v8h*)(dB + 24) = pb3;
            __syncthreads();
            cur ^= 1;
        }
    }

#pragma unroll
    for (int mi = 0; mi < 4; ++mi)
#pragma unroll
        for (int ni = 0; ni < 4; ++ni) {
            int gr0 = gm0 + wm * 64 + mi * 16 + hb;
            int gc  = gn0 + wn * 64 + ni * 16 + ln;
#pragma unroll
            for (int r = 0; r < 8; ++r) {
                long off = (long)(gr0 + r) * GN + gc;
                if (STORE_HALF) Ch[off] = (f16)acc[mi][ni][r];
                else            Cf[off] = acc[mi][ni][r];
            }
        }
}

// ---------------------------------------------------------------- RoPE + repack
__global__ void rope_pack(const f16* __restrict__ Q, const f16* __restrict__ Kq,
                          const f16* __restrict__ V,
                          f16* __restrict__ Qp, f16* __restrict__ Kp, f16* __restrict__ Vt) {
    int idx = blockIdx.x * blockDim.x + threadIdx.x;  // B*S*H*32 = 2^21
    int i = idx & 31;
    int h = (idx >> 5) & 15;
    int s = (idx >> 9) & 2047;
    int b = idx >> 20;
    if (b >= BATCH) return;

    float inv_freq = __powf(10000.0f, -(float)i / 32.0f);
    float ang = (float)s * inv_freq;
    float sn, cs;
    __sincosf(ang, &sn, &cs);

    long row = (long)(b * SEQ + s) * D_MODEL;
    int c1 = h * HD + i, c2 = c1 + 32;
    long pbase = ((long)(b * N_HEADS + h) * SEQ + s) * HD;

    float q1 = (float)Q[row + c1], q2 = (float)Q[row + c2];
    Qp[pbase + i]      = (f16)(q1 * cs - q2 * sn);
    Qp[pbase + 32 + i] = (f16)(q1 * sn + q2 * cs);

    float k1 = (float)Kq[row + c1], k2 = (float)Kq[row + c2];
    Kp[pbase + i]      = (f16)(k1 * cs - k2 * sn);
    Kp[pbase + 32 + i] = (f16)(k1 * sn + k2 * cs);

    long vb = (long)(b * N_HEADS + h) * HD;
    Vt[(vb + i)      * SEQ + s] = V[row + c1];
    Vt[(vb + 32 + i) * SEQ + s] = V[row + c2];
}

// ---------------------------------------------------------------- flash attention
// one wave = one 16-row Q tile of one (b,h); K-step = 32 keys
__global__ __launch_bounds__(256) void attn_fa(const f16* __restrict__ Qp,
                                               const f16* __restrict__ Kp,
                                               const f16* __restrict__ Vt,
                                               f16* __restrict__ Mg) {
    __shared__ f16 Ps[8 * 16 * 40];   // per-wave P tile scratch (stride 40)
    const int t = threadIdx.x;
    const int lane = t & 31, wave = t >> 5;
    const int wg = blockIdx.x * 8 + wave;       // 0..4095
    const int b = wg >> 11;
    const int rem = wg & 2047;
    const int h = rem >> 7;
    const int q0 = (rem & 127) << 4;
    const int ln = lane & 15;
    const int hb = (lane < 16) ? 0 : 8;
    const int kb16 = (lane < 16) ? 0 : 16;
    f16* P = &Ps[wave * 16 * 40];

    const f16* Qbase = Qp + ((long)(b * N_HEADS + h) * SEQ + q0) * HD;
    const f16* Kbase = Kp + (long)(b * N_HEADS + h) * SEQ * HD;
    const f16* Vbase = Vt + (long)(b * N_HEADS + h) * HD * SEQ;

    v16h qa[2];
#pragma unroll
    for (int c = 0; c < 2; ++c) {
        const f16* p = Qbase + ln * HD + c * 32 + kb16;
        qa[c] = mk16(*(const v8h*)p, *(const v8h*)(p + 8));
    }

    float mrun[8], lrun[8];
#pragma unroll
    for (int r = 0; r < 8; ++r) { mrun[r] = -1e30f; lrun[r] = 0.0f; }
    v8f o[4] = {};
    const float scale = 0.03125f; // 1/sqrt(1024)

    for (int kk = 0; kk < SEQ; kk += 32) {
        // scores: 16x32 tile = 2 C-frags, 2 WMMA each
        v8f sc[2];
#pragma unroll
        for (int tt = 0; tt < 2; ++tt) {
            const f16* kb = Kbase + (long)(kk + tt * 16 + ln) * HD;
            v16h k0 = *(const v16h*)(kb + kb16);
            v16h k1 = *(const v16h*)(kb + 32 + kb16);
            v8f z = {};
            z = __builtin_amdgcn_wmma_f32_16x16x32_f16(false, qa[0], false, k0, (short)0, z, false, false);
            z = __builtin_amdgcn_wmma_f32_16x16x32_f16(false, qa[1], false, k1, (short)0, z, false, false);
            sc[tt] = z;
        }

        // V fragments: independent of softmax -> issue loads now so they
        // overlap the exp/shuffle sequence (asm fence below would pin them)
        v16h vb[4];
#pragma unroll
        for (int ni = 0; ni < 4; ++ni) {
            const f16* vp = Vbase + (long)(ni * 16 + ln) * SEQ + kk + kb16;
            vb[ni] = *(const v16h*)vp;
        }

        float alpha[8];
#pragma unroll
        for (int r = 0; r < 8; ++r) {
            float s0 = sc[0][r] * scale, s1 = sc[1][r] * scale;
            float mx = fmaxf(s0, s1);
            mx = fmaxf(mx, __shfl_xor(mx, 1, 32));
            mx = fmaxf(mx, __shfl_xor(mx, 2, 32));
            mx = fmaxf(mx, __shfl_xor(mx, 4, 32));
            mx = fmaxf(mx, __shfl_xor(mx, 8, 32));
            float mnew = fmaxf(mrun[r], mx);
            alpha[r] = __expf(mrun[r] - mnew);
            mrun[r] = mnew;
            float p0 = __expf(s0 - mnew), p1 = __expf(s1 - mnew);
            float rs = p0 + p1;
            rs += __shfl_xor(rs, 1, 32);
            rs += __shfl_xor(rs, 2, 32);
            rs += __shfl_xor(rs, 4, 32);
            rs += __shfl_xor(rs, 8, 32);
            lrun[r] = lrun[r] * alpha[r] + rs;
            P[(r + hb) * 40 + ln]      = (f16)p0;
            P[(r + hb) * 40 + 16 + ln] = (f16)p1;
        }
#pragma unroll
        for (int ni = 0; ni < 4; ++ni)
#pragma unroll
            for (int r = 0; r < 8; ++r) o[ni][r] *= alpha[r];

        asm volatile("s_wait_dscnt 0" ::: "memory");   // wave-local LDS RAW

        const f16* pp = P + ln * 40 + kb16;
        v16h pa = mk16(*(const v8h*)pp, *(const v8h*)(pp + 8));
#pragma unroll
        for (int ni = 0; ni < 4; ++ni)
            o[ni] = __builtin_amdgcn_wmma_f32_16x16x32_f16(false, pa, false, vb[ni], (short)0, o[ni], false, false);
    }

    float inv[8];
#pragma unroll
    for (int r = 0; r < 8; ++r) inv[r] = 1.0f / lrun[r];
    long row0 = (long)(b * SEQ + q0 + hb);
#pragma unroll
    for (int ni = 0; ni < 4; ++ni) {
        int col = h * HD + ni * 16 + ln;
#pragma unroll
        for (int r = 0; r < 8; ++r)
            Mg[(row0 + r) * D_MODEL + col] = (f16)(o[ni][r] * inv[r]);
    }
}

// ---------------------------------------------------------------- launch
extern "C" void kernel_launch(void* const* d_in, const int* in_sizes, int n_in,
                              void* d_out, int out_size, void* d_ws, size_t ws_size,
                              hipStream_t stream) {
    const float* x  = (const float*)d_in[0];
    const float* qw = (const float*)d_in[1];
    const float* kw = (const float*)d_in[2];
    const float* vw = (const float*)d_in[3];
    const float* ow = (const float*)d_in[4];
    float* out = (float*)d_out;

    const size_t MB = 1ull << 20;
    char* ws = (char*)d_ws;
    f16* wqT = (f16*)(ws + 0 * MB);
    f16* wkT = (f16*)(ws + 2 * MB);
    f16* wvT = (f16*)(ws + 4 * MB);
    f16* woT = (f16*)(ws + 6 * MB);
    f16* xf  = (f16*)(ws + 8 * MB);    // 8 MB
    f16* Qf  = (f16*)(ws + 16 * MB);   // 8 MB each
    f16* Kf  = (f16*)(ws + 24 * MB);
    f16* Vf  = (f16*)(ws + 32 * MB);
    f16* Qp  = (f16*)(ws + 40 * MB);
    f16* Kp  = (f16*)(ws + 48 * MB);
    f16* Vt  = (f16*)(ws + 56 * MB);
    f16* Mg  = (f16*)(ws + 64 * MB);   // total 72 MB

    // converts
    cvt_f16_kernel<<<(MROWS * D_MODEL + 255) / 256, 256, 0, stream>>>(x, xf, MROWS * D_MODEL);
    cvt_wT_kernel<<<(D_MODEL * D_MODEL) / 256, 256, 0, stream>>>(qw, wqT);
    cvt_wT_kernel<<<(D_MODEL * D_MODEL) / 256, 256, 0, stream>>>(kw, wkT);
    cvt_wT_kernel<<<(D_MODEL * D_MODEL) / 256, 256, 0, stream>>>(vw, wvT);
    cvt_wT_kernel<<<(D_MODEL * D_MODEL) / 256, 256, 0, stream>>>(ow, woT);

    // projections: [4096,1024] x [1024,1024] -> f16
    dim3 ggrid(GN / 256, MROWS / 128);
    gemm_wmma<1><<<ggrid, 256, 0, stream>>>(xf, wqT, nullptr, Qf);
    gemm_wmma<1><<<ggrid, 256, 0, stream>>>(xf, wkT, nullptr, Kf);
    gemm_wmma<1><<<ggrid, 256, 0, stream>>>(xf, wvT, nullptr, Vf);

    // rope + layout repack
    rope_pack<<<(BATCH * SEQ * N_HEADS * 32) / 256, 256, 0, stream>>>(Qf, Kf, Vf, Qp, Kp, Vt);

    // flash attention: 4096 q-tiles, 8 waves/block
    attn_fa<<<512, 256, 0, stream>>>(Qp, Kp, Vt, Mg);

    // output projection -> f32 d_out
    gemm_wmma<0><<<ggrid, 256, 0, stream>>>(Mg, woT, out, nullptr);
}
